// TransformerDecoder_25735444037818
// MI455X (gfx1250) — matrix operand order repaired and checked by reading, not compiled, not run on hardware
//
#include <hip/hip_runtime.h>

typedef __bf16 bf16;
typedef __attribute__((ext_vector_type(16))) __bf16 v16bf;
typedef __attribute__((ext_vector_type(8)))  __bf16 bf16x8;
typedef __attribute__((ext_vector_type(8)))  float  v8f;
typedef unsigned int u32x4 __attribute__((ext_vector_type(4)));
typedef int i32x4 __attribute__((ext_vector_type(4)));
typedef int i32x8 __attribute__((ext_vector_type(8)));

__device__ __forceinline__ bf16 f2bf(float f) {
  union { float f; unsigned u; } c; c.f = f;
  unsigned r = c.u + 0x7FFFu + ((c.u >> 16) & 1u);
  unsigned short h = (unsigned short)(r >> 16);
  return __builtin_bit_cast(bf16, h);
}

__device__ __forceinline__ v8f wmma_bf16(v16bf a, v16bf b, v8f c) {
  return __builtin_amdgcn_wmma_f32_16x16x32_bf16(false, a, false, b, (short)0, c,
                                                 false, false);
}

#if __has_builtin(__builtin_amdgcn_tensor_load_to_lds) && \
    __has_builtin(__builtin_amdgcn_s_wait_tensorcnt)
#define USE_TDM 1
#if __has_include(<hip/amd_detail/amd_gfx1250_TDM.h>)
#define TDM_6ARG 1
#endif
#endif

#ifdef USE_TDM
// Issue a TDM 2-D tile load Global->LDS per CDNA5 D# (ISA 08 / 8.3-8.6).
// data_size = 2 bytes. LDS padding: 8 DWORDs (16 bf16) inserted after every
// 16 DWORDs (32 bf16) stored -> row stride 48 elements, matching As/Bs below.
// tile_d0 must be 32 elements wide for this pad config.
__device__ __forceinline__ void tdm_load_tile_2d(
    unsigned lds_off, const bf16* gaddr, unsigned tensor_d0, unsigned tensor_d1,
    unsigned stride0_elems, unsigned tile_d0, unsigned tile_d1) {
  unsigned long long ga = (unsigned long long)(size_t)gaddr;
  u32x4 g0 = {};
  g0[0] = 1u;                                   // count=1, user descriptor
  g0[1] = lds_off;                              // LDS byte address
  g0[2] = (unsigned)ga;                         // global_addr[31:0]
  g0[3] = (unsigned)(ga >> 32) | (2u << 30);    // global_addr[56:32] | type=2
  i32x8 g1 = {};
  g1[0] = (int)((1u << 16)    // data_size = 2B
                | (1u << 20)  // pad_enable
                | (3u << 22)  // pad_interval: 16 DWORDs
                | (7u << 25));  // pad_amount: 8 DWORDs
  g1[1] = (int)((tensor_d0 & 0xFFFFu) << 16);
  g1[2] = (int)((tensor_d0 >> 16) | ((tensor_d1 & 0xFFFFu) << 16));
  g1[3] = (int)((tensor_d1 >> 16) | (tile_d0 << 16));
  g1[4] = (int)(tile_d1 & 0xFFFFu);
  g1[5] = (int)stride0_elems;                   // tensor_dim0_stride[31:0]
  g1[6] = 0;
  g1[7] = 0;
  i32x4 g2 = {}, g3 = {};
#ifdef TDM_6ARG
  i32x8 g4 = {};
  __builtin_amdgcn_tensor_load_to_lds(g0, g1, g2, g3, g4, 0);
#else
  __builtin_amdgcn_tensor_load_to_lds(g0, g1, g2, g3, 0);
#endif
}
__device__ __forceinline__ unsigned lds_offset_of(const void* p) {
  return (unsigned)(size_t)p;  // LDS offset lives in the low 32 bits
}
#endif

// ------------------------------------------ transpose + f32->bf16 converter
// src: [L][K][N] f32 row-major; dst: [L][N][K] bf16 row-major (i.e. B^T).
__global__ __launch_bounds__(256) void cvt_transpose_bf16(
    const float* __restrict__ src, bf16* __restrict__ dst, int K, int N) {
  __shared__ float t[32][33];
  const int l = blockIdx.z;
  const int k0 = blockIdx.y * 32, n0 = blockIdx.x * 32;
  const float* s = src + (size_t)l * K * N;
  bf16* d = dst + (size_t)l * K * N;
  const int tx = threadIdx.x & 31, ty = threadIdx.x >> 5;  // 32 x 8
#pragma unroll
  for (int i = 0; i < 32; i += 8)
    t[ty + i][tx] = s[(size_t)(k0 + ty + i) * N + n0 + tx];
  __syncthreads();
#pragma unroll
  for (int i = 0; i < 32; i += 8)
    d[(size_t)(n0 + ty + i) * K + k0 + tx] = f2bf(t[tx][ty + i]);
}

// ----------------------------------------------------------- embedding + PE
__global__ __launch_bounds__(256) void embed_kernel(
    const int* __restrict__ tokens, const float* __restrict__ emb,
    const float* __restrict__ pe, float* __restrict__ x, bf16* __restrict__ xb,
    int S, int Dn) {
  int row = blockIdx.x;  // b*S + s
  int s = row % S;
  int tok = tokens[row];
  const float* er = emb + (size_t)tok * Dn;
  const float* pr = pe + (size_t)s * Dn;
  float* xr = x + (size_t)row * Dn;
  bf16* xbr = xb + (size_t)row * Dn;
  for (int d = threadIdx.x; d < Dn; d += 256) {
    float v = er[d] * 32.0f + pr[d];  // sqrt(1024) = 32
    xr[d] = v;
    xbr[d] = f2bf(v);
  }
}

// -------------------------------------------------------- WMMA bf16 GEMM
// C[M,N] = A[M,K] * Bt^T[K,N] + bias, with Bt given pre-transposed as [N][K].
// BM=128 BN=64 BK=32, 256 thr = 8 waves, wave grid 4(m) x 2(n),
// each wave = 32x32 = 2x2 wmma tiles.
// EPI: 0 fp32 out, 1 bf16 out, 2 relu+bf16 out, 3 bf16 out transposed for
// attention V ([B][H][DK][S] with S=2048, H=16, DK=64).
template <int EPI>
__global__ __launch_bounds__(256) void gemm_bf16_wmma(
    const bf16* __restrict__ A, const bf16* __restrict__ Bt,
    const float* __restrict__ bias, float* __restrict__ Cf,
    bf16* __restrict__ Cb, int M, int N, int K) {
  __shared__ bf16 As[128][48];  // row stride 48 elems = 96B (16B aligned)
  __shared__ bf16 Bs[64][48];   // [n][k], same padded stride
  const int tid = threadIdx.x;
  const int wid = tid >> 5, lane = tid & 31;
  const int wm = wid & 3, wn = wid >> 2;
  const int bm = blockIdx.y * 128, bn = blockIdx.x * 64;
  const int half8 = (lane & 16) ? 8 : 0;   // A row-half K offset
  const int mrow = lane & 15;
  const int kofsB = (lane & 16) ? 16 : 0;  // B lane-half K offset

  v8f acc[2][2] = {};

  const int ar = tid >> 2;        // 0..63
  const int ac = (tid & 3) * 8;   // 0/8/16/24
  const int br = tid >> 2;        // 0..63 (B^T row = n)
  const int bc = (tid & 3) * 8;   // k chunk

  for (int k0 = 0; k0 < K; k0 += 32) {
#ifdef USE_TDM
    // Tensor Data Mover stages both tiles; wave 0 issues and waits, then the
    // barrier publishes them; TENSORcnt is per-wave.
    if (wid == 0) {
      tdm_load_tile_2d(lds_offset_of(&As[0][0]), A + (size_t)bm * K + k0,
                       (unsigned)K, (unsigned)M, (unsigned)K, 32u, 128u);
      tdm_load_tile_2d(lds_offset_of(&Bs[0][0]), Bt + (size_t)bn * K + k0,
                       (unsigned)K, (unsigned)N, (unsigned)K, 32u, 64u);
      __builtin_amdgcn_s_wait_tensorcnt(0);
    }
#else
#pragma unroll
    for (int rr = 0; rr < 128; rr += 64) {
      const bf16* src = A + (size_t)(bm + ar + rr) * K + k0 + ac;
      *(bf16x8*)&As[ar + rr][ac] = *(const bf16x8*)src;
    }
    {
      const bf16* src = Bt + (size_t)(bn + br) * K + k0 + bc;
      *(bf16x8*)&Bs[br][bc] = *(const bf16x8*)src;
    }
#endif
    if (k0 + 32 < K) {  // prefetch next K tile (global_prefetch_b8)
      __builtin_prefetch(A + (size_t)(bm + ar) * K + k0 + 32 + ac, 0, 1);
      __builtin_prefetch(Bt + (size_t)(bn + br) * K + k0 + 32 + bc, 0, 1);
    }
    __syncthreads();

    union { v16bf v; bf16x8 h8[2]; } af[2], bfr[2];
#pragma unroll
    for (int t = 0; t < 2; t++) {
      const bf16* p = &As[wm * 32 + t * 16 + mrow][0];
      af[t].h8[0] = *(const bf16x8*)(p + half8);
      af[t].h8[1] = *(const bf16x8*)(p + 16 + half8);
      const bf16* q = &Bs[wn * 32 + t * 16 + mrow][0];
      bfr[t].h8[0] = *(const bf16x8*)(q + half8);
      bfr[t].h8[1] = *(const bf16x8*)(q + 16 + half8);
    }
#pragma unroll
    for (int tm = 0; tm < 2; tm++)
#pragma unroll
      for (int tn = 0; tn < 2; tn++)
        acc[tm][tn] = wmma_bf16(af[tm].v, bfr[tn].v, acc[tm][tn]);
    __syncthreads();
  }

#pragma unroll
  for (int tm = 0; tm < 2; tm++)
#pragma unroll
    for (int tn = 0; tn < 2; tn++) {
      int col = bn + wn * 32 + tn * 16 + mrow;
      float bv = bias[col];
#pragma unroll
      for (int r = 0; r < 8; r++) {
        int row = bm + wm * 32 + tm * 16 + half8 + r;
        float v = acc[tm][tn][r] + bv;
        if (EPI == 2) v = v > 0.0f ? v : 0.0f;
        if (EPI == 0) {
          Cf[(size_t)row * N + col] = v;
        } else if (EPI == 3) {
          // [B][H][DK][S]: b=row>>11, s=row&2047, h=col>>6, dk=col&63
          size_t idx =
              (((size_t)(row >> 11) * 16 + (col >> 6)) * 64 + (col & 63)) *
                  2048 +
              (row & 2047);
          Cb[idx] = f2bf(v);
        } else {
          Cb[(size_t)row * N + col] = f2bf(v);
        }
      }
    }
}

// ------------------------------------------------ flash attention (causal)
// grid = (S/128, H, B); block = 256 (8 waves, 16 query rows each).
// Vt is pre-transposed: [B][H][DK][S].
__global__ __launch_bounds__(256) void attn_flash(
    const bf16* __restrict__ Q, const bf16* __restrict__ Kb,
    const bf16* __restrict__ Vt, bf16* __restrict__ Ctx) {
  const int S = 2048, Dm = 1024, DK = 64;
  __shared__ bf16 Ks[64][80];     // [keypos][dk], stride 160B
  __shared__ bf16 Vs[64][80];     // [dk][keypos]
  __shared__ bf16 Ps[8][16][80];  // per-wave P scratch [qrow][keypos]
  const int tid = threadIdx.x, wid = tid >> 5, lane = tid & 31;
  const int half8 = (lane & 16) ? 8 : 0;
  const int mrow = lane & 15;
  const int kofsB = (lane & 16) ? 16 : 0;
  const int b = blockIdx.z, h = blockIdx.y;
  const int q0 = blockIdx.x * 128;

  // Q fragments straight from global
  const bf16* qrow = Q + (size_t)(b * S + q0 + wid * 16 + mrow) * Dm + h * DK;
  union { v16bf v; bf16x8 h8[2]; } qa[2];
#pragma unroll
  for (int t = 0; t < 2; t++) {
    qa[t].h8[0] = *(const bf16x8*)(qrow + t * 32 + half8);
    qa[t].h8[1] = *(const bf16x8*)(qrow + t * 32 + 16 + half8);
  }

  float mstat[8], lstat[8];
#pragma unroll
  for (int r = 0; r < 8; r++) { mstat[r] = -1e30f; lstat[r] = 0.0f; }
  v8f oacc[4] = {};

  const int lr = tid >> 3;        // 0..31
  const int lc = (tid & 7) * 8;   // 0..56
  const bf16* vtg = Vt + ((size_t)b * 16 + h) * 64 * 2048;  // [dk][s]

  const int kvEnd = q0 + 128;
  for (int kv0 = 0; kv0 < kvEnd; kv0 += 64) {
    // stage K tile [keypos][dk] and V^T tile [dk][keypos]: vector copies only
#pragma unroll
    for (int rr = 0; rr < 64; rr += 32) {
      const bf16* ksrc = Kb + (size_t)(b * S + kv0 + lr + rr) * Dm + h * DK + lc;
      *(bf16x8*)&Ks[lr + rr][lc] = *(const bf16x8*)ksrc;
      const bf16* vsrc = vtg + (size_t)(lr + rr) * 2048 + kv0 + lc;
      *(bf16x8*)&Vs[lr + rr][lc] = *(const bf16x8*)vsrc;
    }
    __syncthreads();

    // scores S = Q K^T : 4 key tiles of 16, contraction over dk (2 steps)
    v8f sacc[4];
#pragma unroll
    for (int nt = 0; nt < 4; nt++) {
      const bf16* kp = &Ks[nt * 16 + mrow][0];
      union { v16bf v; bf16x8 h8[2]; } b0, b1;
      b0.h8[0] = *(const bf16x8*)(kp + half8);
      b0.h8[1] = *(const bf16x8*)(kp + 16 + half8);
      b1.h8[0] = *(const bf16x8*)(kp + 32 + half8);
      b1.h8[1] = *(const bf16x8*)(kp + 48 + half8);
      v8f z = {};
      z = wmma_bf16(qa[0].v, b0.v, z);
      z = wmma_bf16(qa[1].v, b1.v, z);
      sacc[nt] = z;
    }

    // online softmax per query row (rows live in lane halves)
#pragma unroll
    for (int r = 0; r < 8; r++) {
      int qg = q0 + wid * 16 + half8 + r;
      float sv[4];
      float mloc = -1e30f;
#pragma unroll
      for (int nt = 0; nt < 4; nt++) {
        int kc = kv0 + nt * 16 + mrow;
        float s = sacc[nt][r] * 0.125f;  // 1/sqrt(64)
        if (kc > qg) s = -1e30f;
        sv[nt] = s;
        mloc = fmaxf(mloc, s);
      }
#pragma unroll
      for (int m = 8; m >= 1; m >>= 1) mloc = fmaxf(mloc, __shfl_xor(mloc, m, 16));
      float mn = fmaxf(mstat[r], mloc);
      float corr = __expf(mstat[r] - mn);
      float psum = 0.0f;
#pragma unroll
      for (int nt = 0; nt < 4; nt++) {
        float p = __expf(sv[nt] - mn);
        psum += p;
        Ps[wid][half8 + r][nt * 16 + mrow] = f2bf(p);
      }
#pragma unroll
      for (int m = 8; m >= 1; m >>= 1) psum += __shfl_xor(psum, m, 16);
      lstat[r] = lstat[r] * corr + psum;
      mstat[r] = mn;
#pragma unroll
      for (int nt = 0; nt < 4; nt++) oacc[nt][r] *= corr;
    }

    // O += P V : A = P (via per-wave LDS transpose), B = V^T tile
    union { v16bf v; bf16x8 h8[2]; } pa0, pa1;
    const bf16* pp = &Ps[wid][mrow][0];
    pa0.h8[0] = *(const bf16x8*)(pp + half8);
    pa0.h8[1] = *(const bf16x8*)(pp + 16 + half8);
    pa1.h8[0] = *(const bf16x8*)(pp + 32 + half8);
    pa1.h8[1] = *(const bf16x8*)(pp + 48 + half8);
#pragma unroll
    for (int nt = 0; nt < 4; nt++) {
      const bf16* vp = &Vs[nt * 16 + mrow][0];
      union { v16bf v; bf16x8 h8[2]; } v0, v1;
      v0.h8[0] = *(const bf16x8*)(vp + half8);
      v0.h8[1] = *(const bf16x8*)(vp + 16 + half8);
      v1.h8[0] = *(const bf16x8*)(vp + 32 + half8);
      v1.h8[1] = *(const bf16x8*)(vp + 48 + half8);
      oacc[nt] = wmma_bf16(pa0.v, v0.v, oacc[nt]);
      oacc[nt] = wmma_bf16(pa1.v, v1.v, oacc[nt]);
    }
    __syncthreads();
  }

#pragma unroll
  for (int nt = 0; nt < 4; nt++)
#pragma unroll
    for (int r = 0; r < 8; r++) {
      int row = b * S + q0 + wid * 16 + half8 + r;
      int col = h * DK + nt * 16 + mrow;
      Ctx[(size_t)row * Dm + col] = f2bf(oacc[nt][r] / lstat[r]);
    }
}

// ------------------------------------------------ residual add + layernorm
__global__ __launch_bounds__(256) void add_ln_kernel(
    float* __restrict__ x, const float* __restrict__ delta,
    const float* __restrict__ g, const float* __restrict__ beta,
    bf16* __restrict__ xb) {
  const int Dn = 1024;
  int row = blockIdx.x;
  float* xr = x + (size_t)row * Dn;
  const float* dr = delta + (size_t)row * Dn;
  __shared__ float s1[256], s2[256];
  float loc[4];
  float a = 0.0f, sq = 0.0f;
#pragma unroll
  for (int i = 0; i < 4; i++) {
    int d = threadIdx.x + i * 256;
    float v = xr[d] + dr[d];
    loc[i] = v;
    a += v;
    sq += v * v;
  }
  s1[threadIdx.x] = a;
  s2[threadIdx.x] = sq;
  __syncthreads();
  for (int o = 128; o > 0; o >>= 1) {
    if ((int)threadIdx.x < o) {
      s1[threadIdx.x] += s1[threadIdx.x + o];
      s2[threadIdx.x] += s2[threadIdx.x + o];
    }
    __syncthreads();
  }
  float mu = s1[0] * (1.0f / 1024.0f);
  float var = s2[0] * (1.0f / 1024.0f) - mu * mu;
  float rs = rsqrtf(var + 1e-5f);
#pragma unroll
  for (int i = 0; i < 4; i++) {
    int d = threadIdx.x + i * 256;
    float y = (loc[i] - mu) * rs * g[d] + beta[d];
    xr[d] = y;
    xb[(size_t)row * Dn + d] = f2bf(y);
  }
}

// ------------------------------------------------------------------- launch
extern "C" void kernel_launch(void* const* d_in, const int* in_sizes, int n_in,
                              void* d_out, int out_size, void* d_ws,
                              size_t ws_size, hipStream_t stream) {
  const int* tokens = (const int*)d_in[0];
  const float* pe = (const float*)d_in[1];
  const float* embedW = (const float*)d_in[2];
  const float* Wq = (const float*)d_in[3], * bq = (const float*)d_in[4];
  const float* Wk = (const float*)d_in[5], * bk = (const float*)d_in[6];
  const float* Wv = (const float*)d_in[7], * bv = (const float*)d_in[8];
  const float* Wo = (const float*)d_in[9], * bo = (const float*)d_in[10];
  const float* W1 = (const float*)d_in[11], * b1 = (const float*)d_in[12];
  const float* W2 = (const float*)d_in[13], * b2 = (const float*)d_in[14];
  const float* ln1g = (const float*)d_in[15], * ln1b = (const float*)d_in[16];
  const float* ln2g = (const float*)d_in[17], * ln2b = (const float*)d_in[18];
  const float* Wout = (const float*)d_in[19], * bout = (const float*)d_in[20];

  const int Bn = 4, S = 2048, D = 1024, H = 16, Ln = 6, F = 4096, V = 32000;
  const int M = Bn * S;  // 8192

  char* w = (char*)d_ws;
  size_t off = 0;
  auto take = [&](size_t bytes) -> void* {
    void* p = w + off;
    off += (bytes + 255) & ~(size_t)255;
    return p;
  };
  // all weights stored TRANSPOSED ([N][K]) in bf16
  bf16* wqb = (bf16*)take((size_t)Ln * D * D * 2);
  bf16* wkb = (bf16*)take((size_t)Ln * D * D * 2);
  bf16* wvb = (bf16*)take((size_t)Ln * D * D * 2);
  bf16* wob = (bf16*)take((size_t)Ln * D * D * 2);
  bf16* w1b = (bf16*)take((size_t)Ln * D * F * 2);
  bf16* w2b = (bf16*)take((size_t)Ln * F * D * 2);
  bf16* woutb = (bf16*)take((size_t)D * V * 2);
  float* x = (float*)take((size_t)M * D * 4);
  bf16* xb = (bf16*)take((size_t)M * D * 2);
  bf16* qb = (bf16*)take((size_t)M * D * 2);
  bf16* kbuf = (bf16*)take((size_t)M * D * 2);
  bf16* vtb = (bf16*)take((size_t)M * D * 2);   // [B][H][DK][S]
  bf16* ctxb = (bf16*)take((size_t)M * D * 2);
  bf16* hb = (bf16*)take((size_t)M * F * 2);
  float* tmp = (float*)take((size_t)M * D * 4);

  dim3 tDD(D / 32, D / 32, Ln);
  cvt_transpose_bf16<<<tDD, 256, 0, stream>>>(Wq, wqb, D, D);
  cvt_transpose_bf16<<<tDD, 256, 0, stream>>>(Wk, wkb, D, D);
  cvt_transpose_bf16<<<tDD, 256, 0, stream>>>(Wv, wvb, D, D);
  cvt_transpose_bf16<<<tDD, 256, 0, stream>>>(Wo, wob, D, D);
  cvt_transpose_bf16<<<dim3(F / 32, D / 32, Ln), 256, 0, stream>>>(W1, w1b, D, F);
  cvt_transpose_bf16<<<dim3(D / 32, F / 32, Ln), 256, 0, stream>>>(W2, w2b, F, D);
  cvt_transpose_bf16<<<dim3(V / 32, D / 32, 1), 256, 0, stream>>>(Wout, woutb, D, V);

  embed_kernel<<<M, 256, 0, stream>>>(tokens, embedW, pe, x, xb, S, D);

  dim3 gD(D / 64, M / 128);
  dim3 gF(F / 64, M / 128);
  dim3 gV(V / 64, M / 128);
  dim3 gA(S / 128, H, Bn);

  for (int l = 0; l < Ln; l++) {
    gemm_bf16_wmma<1><<<gD, 256, 0, stream>>>(xb, wqb + (size_t)l * D * D,
                                              bq + (size_t)l * D, nullptr, qb,
                                              M, D, D);
    gemm_bf16_wmma<1><<<gD, 256, 0, stream>>>(xb, wkb + (size_t)l * D * D,
                                              bk + (size_t)l * D, nullptr, kbuf,
                                              M, D, D);
    gemm_bf16_wmma<3><<<gD, 256, 0, stream>>>(xb, wvb + (size_t)l * D * D,
                                              bv + (size_t)l * D, nullptr, vtb,
                                              M, D, D);
    attn_flash<<<gA, 256, 0, stream>>>(qb, kbuf, vtb, ctxb);
    gemm_bf16_wmma<0><<<gD, 256, 0, stream>>>(ctxb, wob + (size_t)l * D * D,
                                              bo + (size_t)l * D, tmp, nullptr,
                                              M, D, D);
    add_ln_kernel<<<M, 256, 0, stream>>>(x, tmp, ln1g + (size_t)l * D,
                                         ln1b + (size_t)l * D, xb);
    gemm_bf16_wmma<2><<<gF, 256, 0, stream>>>(xb, w1b + (size_t)l * D * F,
                                              b1 + (size_t)l * F, nullptr, hb,
                                              M, F, D);
    gemm_bf16_wmma<0><<<gD, 256, 0, stream>>>(hb, w2b + (size_t)l * F * D,
                                              b2 + (size_t)l * D, tmp, nullptr,
                                              M, D, F);
    add_ln_kernel<<<M, 256, 0, stream>>>(x, tmp, ln2g + (size_t)l * D,
                                         ln2b + (size_t)l * D, xb);
  }
  gemm_bf16_wmma<0><<<gV, 256, 0, stream>>>(xb, woutb, bout, (float*)d_out,
                                            nullptr, M, V, D);
}
